// Attention_7069516169289
// MI455X (gfx1250) — matrix operand order repaired and checked by reading, not compiled, not run on hardware
//
#include <hip/hip_runtime.h>

// ---------------------------------------------------------------------------
// Types for CDNA5 WMMA (gfx1250, wave32)
// ---------------------------------------------------------------------------
typedef __bf16 bf16;
typedef __attribute__((ext_vector_type(16))) __bf16 bf16x16;
typedef __attribute__((ext_vector_type(8)))  __bf16 bf16x8;
typedef __attribute__((ext_vector_type(2)))  __bf16 bf16x2;
typedef __attribute__((ext_vector_type(8)))  float  f32x8;
typedef __attribute__((ext_vector_type(4)))  int    v4i;

#define WMMA_BF16(a, b, c) \
  __builtin_amdgcn_wmma_f32_16x16x32_bf16(false, (a), false, (b), (short)0, (c), false, false)

#define CAT8(lo, hi) __builtin_shufflevector((lo), (hi), 0,1,2,3,4,5,6,7,8,9,10,11,12,13,14,15)

// Problem constants (from the reference)
constexpr int BATCH  = 4;
constexpr int NSEQ   = 2048;
constexpr int DIM    = 1024;
constexpr int HEADS  = 16;
constexpr int DHEAD  = 64;
constexpr int FOUT   = 3 * 1024;
constexpr float SCALE = 0.125f;           // DHEAD^-0.5

// ---------------------------------------------------------------------------
// CDNA5 async global->LDS copy (ASYNCcnt path), with safe fallback
// Builtin signature (from hipcc diagnostic): (v4i AS1*, v4i AS3*, Imm, Imm)
// ---------------------------------------------------------------------------
#if __has_builtin(__builtin_amdgcn_global_load_async_to_lds_b128)
#define HAS_ASYNC_LDS 1
#else
#define HAS_ASYNC_LDS 0
#endif

__device__ __forceinline__ void async_copy16(const void* g, void* l) {
#if HAS_ASYNC_LDS
  __builtin_amdgcn_global_load_async_to_lds_b128(
      (__attribute__((address_space(1))) v4i*)g,
      (__attribute__((address_space(3))) v4i*)l, 0, 0);
#else
  *(uint4*)l = *(const uint4*)g;          // synchronous fallback
#endif
}

__device__ __forceinline__ void wait_async_older4_done() {
#if HAS_ASYNC_LDS
  asm volatile("s_wait_asynccnt 0x4" ::: "memory");   // oldest 4 of <=8 done
#else
  asm volatile("s_wait_dscnt 0x0" ::: "memory");
#endif
}

// ---------------------------------------------------------------------------
// Kernel 1: QKV projection.  C[8192,3072] = X[8192,1024] * W[1024,3072]
// 128x128 tile, K-step 32, 8 waves; software-pipelined global loads;
// epilogue scatters bf16 Q (pre-scaled), K [b,h,n,d], V^T [b,h,d,n].
// ---------------------------------------------------------------------------
__global__ __launch_bounds__(256) void qkv_gemm_kernel(
    const float* __restrict__ x, const float* __restrict__ w,
    bf16* __restrict__ q_ws, bf16* __restrict__ k_ws, bf16* __restrict__ vt_ws)
{
  __shared__ __align__(16) bf16 As[128][40];   // X tile  [m][k], pad->40
  __shared__ __align__(16) bf16 Bs[128][40];   // W tile transposed [n][k]

  const int tid  = threadIdx.x;
  const int lane = tid & 31;
  const int wv   = tid >> 5;
  const int wr   = wv & 3;                 // 4 row groups of 32
  const int wc   = wv >> 2;                // 2 col groups of 64
  const int l16  = lane & 15;
  const int m0   = blockIdx.x * 128;
  const int n0   = blockIdx.y * 128;
  const int kb_a = (lane < 16) ? 0 : 8;
  const int kb_b = (lane < 16) ? 0 : 8;

  // Per-thread staging assignments (constant across K loop)
  const int arow  = tid >> 1;              // A: 128 rows, 2 threads/row
  const int akoff = (tid & 1) * 16;        // 16 floats each
  const int bnc   = tid >> 4;              // B: 16 n-chunks of 8
  const int bkp   = (tid & 15) * 2;        // k-pair base (even)

  float a_st[16], b0_st[8], b1_st[8];

  auto load_stage = [&](int k0) {
    const float4* pa =
        reinterpret_cast<const float4*>(x + (size_t)(m0 + arow) * DIM + k0 + akoff);
#pragma unroll
    for (int i = 0; i < 4; ++i) {
      float4 t = pa[i];
      a_st[4*i+0] = t.x; a_st[4*i+1] = t.y; a_st[4*i+2] = t.z; a_st[4*i+3] = t.w;
    }
    const float4* pb0 =
        reinterpret_cast<const float4*>(w + (size_t)(k0 + bkp) * FOUT + n0 + bnc * 8);
    const float4* pb1 =
        reinterpret_cast<const float4*>(w + (size_t)(k0 + bkp + 1) * FOUT + n0 + bnc * 8);
#pragma unroll
    for (int i = 0; i < 2; ++i) {
      float4 t0 = pb0[i], t1 = pb1[i];
      b0_st[4*i+0] = t0.x; b0_st[4*i+1] = t0.y; b0_st[4*i+2] = t0.z; b0_st[4*i+3] = t0.w;
      b1_st[4*i+0] = t1.x; b1_st[4*i+1] = t1.y; b1_st[4*i+2] = t1.z; b1_st[4*i+3] = t1.w;
    }
  };

  f32x8 acc[2][4] = {};
  load_stage(0);

  for (int k0 = 0; k0 < DIM; k0 += 32) {
    // ---- commit staged tile to LDS (bf16) ----
    {
      bf16x8 lo, hi;
#pragma unroll
      for (int i = 0; i < 8; ++i) { lo[i] = (bf16)a_st[i]; hi[i] = (bf16)a_st[8 + i]; }
      *reinterpret_cast<bf16x8*>(&As[arow][akoff])     = lo;
      *reinterpret_cast<bf16x8*>(&As[arow][akoff + 8]) = hi;
#pragma unroll
      for (int i = 0; i < 8; ++i) {
        bf16x2 pr; pr[0] = (bf16)b0_st[i]; pr[1] = (bf16)b1_st[i];
        *reinterpret_cast<bf16x2*>(&Bs[bnc * 8 + i][bkp]) = pr;   // ds_store_b32
      }
    }
    __syncthreads();

    // ---- prefetch next K-step while WMMAs run ----
    if (k0 + 32 < DIM) load_stage(k0 + 32);

    // ---- fragments + 8 WMMAs ----
    bf16x16 afrag[2], bfrag[4];
#pragma unroll
    for (int mt = 0; mt < 2; ++mt) {
      const int row = wr * 32 + mt * 16 + l16;
      bf16x8 lo = *reinterpret_cast<const bf16x8*>(&As[row][kb_a]);
      bf16x8 hi = *reinterpret_cast<const bf16x8*>(&As[row][16 + kb_a]);
      afrag[mt] = CAT8(lo, hi);
    }
#pragma unroll
    for (int nt = 0; nt < 4; ++nt) {
      const int col = wc * 64 + nt * 16 + l16;
      bf16x8 lo = *reinterpret_cast<const bf16x8*>(&Bs[col][kb_b]);
      bf16x8 hi = *reinterpret_cast<const bf16x8*>(&Bs[col][kb_b + 16]);
      bfrag[nt] = CAT8(lo, hi);
    }
#pragma unroll
    for (int mt = 0; mt < 2; ++mt)
#pragma unroll
      for (int nt = 0; nt < 4; ++nt)
        acc[mt][nt] = WMMA_BF16(afrag[mt], bfrag[nt], acc[mt][nt]);

    __syncthreads();
  }

  // ---- epilogue: scalarize wave-uniform values for s_cbranch dispatch ----
  const int radd = (lane < 16) ? 0 : 8;
#pragma unroll
  for (int nt = 0; nt < 4; ++nt) {
    const int fb    = __builtin_amdgcn_readfirstlane(n0 + wc * 64 + nt * 16);
    const int which = fb >> 10;                  // 0=Q 1=K 2=V (scalar)
    const int h     = (fb & 1023) >> 6;          // scalar
    const int d     = (fb & 63) + l16;           // vector (lane)
#pragma unroll
    for (int mt = 0; mt < 2; ++mt) {
      const int mb    = __builtin_amdgcn_readfirstlane(m0 + wr * 32 + mt * 16);
      const int bi    = mb >> 11;                // scalar (tile never straddles)
      const int nbase = mb & 2047;               // scalar
      const size_t bh = (size_t)bi * HEADS + h;
      f32x8 a = acc[mt][nt];
      if (which == 0) {
        bf16* p = q_ws + (bh * NSEQ + nbase + radd) * DHEAD + d;
#pragma unroll
        for (int r = 0; r < 8; ++r) p[r * DHEAD] = (bf16)(a[r] * SCALE);
      } else if (which == 1) {
        bf16* p = k_ws + (bh * NSEQ + nbase + radd) * DHEAD + d;
#pragma unroll
        for (int r = 0; r < 8; ++r) p[r * DHEAD] = (bf16)a[r];
      } else {
        bf16* p = vt_ws + (bh * DHEAD + d) * NSEQ + nbase + radd;
#pragma unroll
        for (int r = 0; r < 8; ++r) p[r] = (bf16)a[r];
      }
    }
  }
}

// ---------------------------------------------------------------------------
// Kernel 2: flash attention.  One WG = one (b,h) x 64 query rows, 4 waves.
// K/V blocks staged once per WG via async global->LDS copies, double-buffered
// (issue block i+1 while computing block i; s_wait_asynccnt + barrier).
// ---------------------------------------------------------------------------
__global__ __launch_bounds__(128) void flash_attn_kernel(
    const bf16* __restrict__ q_ws, const bf16* __restrict__ k_ws,
    const bf16* __restrict__ vt_ws, float* __restrict__ out)
{
  __shared__ __align__(16) bf16 Ks[2][32][72];   // keys x d   (pad 64->72)
  __shared__ __align__(16) bf16 Vs[2][64][40];   // d    x keys(pad 32->40)
  __shared__ __align__(16) bf16 plds[4][16][40]; // per-wave P staging

  const int tid  = threadIdx.x;
  const int lane = tid & 31;
  const int wv   = tid >> 5;
  const int l16  = lane & 15;
  const int qb   = blockIdx.x & 31;
  const int bh   = blockIdx.x >> 5;
  const int bi   = bh >> 4;
  const int h    = bh & 15;
  const int kb_a = (lane < 16) ? 0 : 8;
  const int kb_b = (lane < 16) ? 0 : 8;

  // cooperative async staging of one 32-key K/V block
  auto issue_kv = [&](int j, bf16 (*Ksb)[72], bf16 (*Vsb)[40]) {
    const int krow  = tid >> 3;              // 0..15
    const int kboff = (tid & 7) * 16;        // bytes within 128B row
#pragma unroll
    for (int t = 0; t < 2; ++t) {
      const int r = t * 16 + krow;
      const char* g =
          (const char*)(k_ws + ((size_t)bh * NSEQ + j + r) * DHEAD) + kboff;
      async_copy16(g, (char*)(&Ksb[r][0]) + kboff);
    }
    const int vrow  = tid >> 2;              // 0..31
    const int vboff = (tid & 3) * 16;        // bytes within 64B row
#pragma unroll
    for (int t = 0; t < 2; ++t) {
      const int r = t * 32 + vrow;
      const char* g =
          (const char*)(vt_ws + ((size_t)bh * DHEAD + r) * NSEQ + j) + vboff;
      async_copy16(g, (char*)(&Vsb[r][0]) + vboff);
    }
  };

  // persistent Q fragments
  bf16x16 qf[2];
  {
    const bf16* qp = q_ws + ((size_t)bh * NSEQ + qb * 64 + wv * 16 + l16) * DHEAD;
#pragma unroll
    for (int c = 0; c < 2; ++c) {
      bf16x8 lo = *reinterpret_cast<const bf16x8*>(qp + c * 32 + kb_a);
      bf16x8 hi = *reinterpret_cast<const bf16x8*>(qp + c * 32 + 16 + kb_a);
      qf[c] = CAT8(lo, hi);
    }
  }

  f32x8 oacc[4] = {};
  float mrun[8], lrun[8];
#pragma unroll
  for (int r = 0; r < 8; ++r) { mrun[r] = -3.0e38f; lrun[r] = 0.0f; }

  constexpr int NB = NSEQ / 32;              // 64 key blocks
  issue_kv(0, Ks[0], Vs[0]);

  for (int i = 0; i < NB; ++i) {
    const int jn = (i + 1 < NB) ? (i + 1) * 32 : 0;    // wrap: keeps wait uniform
    issue_kv(jn, Ks[(i + 1) & 1], Vs[(i + 1) & 1]);
    wait_async_older4_done();
    __syncthreads();                         // all waves' copies for block i landed

    bf16 (*Ksb)[72] = Ks[i & 1];
    bf16 (*Vsb)[40] = Vs[i & 1];

    // ---- S = Q * K^T from LDS (scale pre-folded into Q) ----
    f32x8 s[2] = {};
#pragma unroll
    for (int t = 0; t < 2; ++t) {
#pragma unroll
      for (int c = 0; c < 2; ++c) {
        bf16x8 lo = *reinterpret_cast<const bf16x8*>(&Ksb[t * 16 + l16][c * 32 + kb_b]);
        bf16x8 hi = *reinterpret_cast<const bf16x8*>(&Ksb[t * 16 + l16][c * 32 + kb_b + 16]);
        bf16x16 kf = CAT8(lo, hi);
        s[t] = WMMA_BF16(qf[c], kf, s[t]);
      }
    }

    // ---- online softmax (row stats via 16-lane butterflies) ----
#pragma unroll
    for (int r = 0; r < 8; ++r) {
      float a0 = s[0][r], a1 = s[1][r];
      float mloc = fmaxf(a0, a1);
#pragma unroll
      for (int off = 1; off < 16; off <<= 1)
        mloc = fmaxf(mloc, __shfl_xor(mloc, off, 32));
      const float mn    = fmaxf(mrun[r], mloc);
      const float alpha = __expf(mrun[r] - mn);
      const float p0 = __expf(a0 - mn);
      const float p1 = __expf(a1 - mn);
      float ls = p0 + p1;
#pragma unroll
      for (int off = 1; off < 16; off <<= 1)
        ls += __shfl_xor(ls, off, 32);
      lrun[r] = lrun[r] * alpha + ls;
      mrun[r] = mn;
#pragma unroll
      for (int c = 0; c < 4; ++c) oacc[c][r] *= alpha;
      const int rowm = (lane < 16) ? r : r + 8;
      plds[wv][rowm][l16]      = (bf16)p0;
      plds[wv][rowm][16 + l16] = (bf16)p1;
    }
    asm volatile("s_wait_dscnt 0x0" ::: "memory");   // P stores -> reads (same wave)

    // ---- P (C-layout) -> A-fragment via LDS ----
    bf16x8 plo = *reinterpret_cast<const bf16x8*>(&plds[wv][l16][kb_a]);
    bf16x8 phi = *reinterpret_cast<const bf16x8*>(&plds[wv][l16][16 + kb_a]);
    bf16x16 pf = CAT8(plo, phi);

    // ---- O += P * V from LDS ----
#pragma unroll
    for (int c = 0; c < 4; ++c) {
      bf16x8 lo = *reinterpret_cast<const bf16x8*>(&Vsb[c * 16 + l16][kb_b]);
      bf16x8 hi = *reinterpret_cast<const bf16x8*>(&Vsb[c * 16 + l16][kb_b + 16]);
      bf16x16 vf = CAT8(lo, hi);
      oacc[c] = WMMA_BF16(pf, vf, oacc[c]);
    }
    __syncthreads();                         // block i fully consumed
  }

  // ---- normalize and write out[b, n, h*64 + d] (f32) ----
#pragma unroll
  for (int c = 0; c < 4; ++c) {
    const int d = c * 16 + l16;
#pragma unroll
    for (int r = 0; r < 8; ++r) {
      const int rowm = (lane < 16) ? r : r + 8;
      const int n = qb * 64 + wv * 16 + rowm;
      out[((size_t)bi * NSEQ + n) * DIM + h * DHEAD + d] = oacc[c][r] / lrun[r];
    }
  }
}

// ---------------------------------------------------------------------------
// Host entry
// ---------------------------------------------------------------------------
extern "C" void kernel_launch(void* const* d_in, const int* in_sizes, int n_in,
                              void* d_out, int out_size, void* d_ws, size_t ws_size,
                              hipStream_t stream) {
  (void)in_sizes; (void)n_in; (void)out_size; (void)ws_size;
  const float* x = (const float*)d_in[0];        // [4, 2048, 1024]
  const float* w = (const float*)d_in[1];        // [1024, 3072]
  float* out = (float*)d_out;                    // [4, 2048, 1024]

  const size_t per = (size_t)BATCH * HEADS * NSEQ * DHEAD;  // 8,388,608 elems
  bf16* q_ws  = (bf16*)d_ws;
  bf16* k_ws  = q_ws + per;
  bf16* vt_ws = k_ws + per;                      // 48 MB of workspace

  dim3 gGemm(64, 24);                            // 8192/128 x 3072/128
  qkv_gemm_kernel<<<gGemm, 256, 0, stream>>>(x, w, q_ws, k_ws, vt_ws);

  dim3 gAttn(BATCH * HEADS * (NSEQ / 64));       // 2048 workgroups
  flash_attn_kernel<<<gAttn, 128, 0, stream>>>(q_ws, k_ws, vt_ws, out);
}
